// Model_75084618269377
// MI455X (gfx1250) — matrix operand order repaired and checked by reading, compiled, not run on hardware
//
#include <hip/hip_runtime.h>
#include <hip/hip_bf16.h>
#include <math.h>

// ---------------------------------------------------------------------------
// MIM predictive RNN forward for gfx1250 (wave32, WMMA f16 -> f32).
// Convs: implicit GEMM on v_wmma_f32_16x16x32_f16, compile-time shapes.
// One Cout-tile per wave (tile id = blockIdx.y*8 + wave) so nothing gets
// LICM-hoisted; A stream pipelined at distance 2, B stream at distance 1.
// ---------------------------------------------------------------------------

typedef __attribute__((ext_vector_type(16))) _Float16 v16h;
typedef __attribute__((ext_vector_type(8)))  _Float16 v8h;
typedef __attribute__((ext_vector_type(8)))  float    v8f;

#define HW   32
#define PIX  1024          // 32*32
#define NB   2             // batch
#define T64  (NB*64*PIX)   // elements of one 64-channel tensor (131072)

// ------------------------- weight repack: f32 -> f16 [tap][co][cinPad] -----
__global__ void k_convert_w(const float* __restrict__ src, _Float16* __restrict__ dst,
                            int Cout, int Cin, int CinPad, int taps)
{
    int i = blockIdx.x * blockDim.x + threadIdx.x;
    int total = taps * Cout * CinPad;
    if (i >= total) return;
    int ci  = i % CinPad;
    int t2  = i / CinPad;
    int co  = t2 % Cout;
    int tap = t2 / Cout;
    float v = 0.0f;
    if (ci < Cin) v = src[((size_t)co * Cin + ci) * taps + tap];
    dst[i] = (_Float16)v;
}

// ------------------------- implicit-GEMM conv via WMMA ---------------------
// in : [B][CIN][32][32] f32, wgt: [taps][COUT][CINPAD] f16, out: [B][COUT][32][32]
// grid = (64 rows, ceil(MT/8)); block = 256 threads = 8 waves.
// Each block computes one full 32-pixel row; each wave owns ONE Cout 16-tile
// and two N-spans (x 0-15, 16-31) sharing every A fetch -> 32 FLOP/A-byte.
template<int CIN, int CINPAD, int COUT, int K, int PAD>
__global__ __launch_bounds__(256) void k_conv(
    const float* __restrict__ in, const _Float16* __restrict__ wgt,
    float* __restrict__ out)
{
    constexpr int LDSW  = 32 + K - 1;            // 36 (K=5) or 32 (K=1)
    constexpr int STAGE = K * LDSW * CINPAD;
    constexpr int PERCH = K * LDSW;              // halo pixels per channel
    constexpr int MT    = COUT / 16;
    constexpr int NCH   = CINPAD / 32;           // k-chunks per tap
    constexpr int NIT   = K * K * NCH;           // total wmma pairs
    constexpr size_t APITCH = (size_t)COUT * CINPAD;
    __shared__ _Float16 smem[STAGE];

    const int s   = blockIdx.x;                  // 0..63: (b, y)
    const int b   = s >> 5;
    const int y   = s & 31;
    const int tid = threadIdx.x;

    // Stage input halo (f32 -> f16) into LDS; adjacent threads read adjacent gx.
    for (int idx = tid; idx < STAGE; idx += 256) {
        int ci  = idx / PERCH;
        int rm  = idx - ci * PERCH;
        int r   = rm / LDSW;
        int cx  = rm - r * LDSW;
        int gyy = y - PAD + r;
        int gxx = cx - PAD;
        float v = 0.0f;
        if (ci < CIN && (unsigned)gyy < HW && (unsigned)gxx < HW)
            v = in[(((size_t)b * CIN + ci) * HW + gyy) * HW + gxx];
        smem[(r * LDSW + cx) * CINPAD + ci] = (_Float16)v;
    }
    __syncthreads();

    const int wave = tid >> 5;
    const int lane = tid & 31;
    const int lm   = lane & 15;   // n (pixel) / m (cout) index
    const int lh   = lane >> 4;   // half-wave group
    const int mt   = blockIdx.y * 8 + wave;      // this wave's Cout tile
    if (mt >= MT) return;                        // wave-uniform exit

    // B base: n = lm, K = k0 + lh*16 + h (contiguous k order, same as A).
    const _Float16* bbase = smem + lm * CINPAD + lh * 16;
    // A base: row co = mt*16 + lm; K = k0 + lh*16 + {0..15}.
    const _Float16* abase = wgt + ((size_t)mt * 16 + lm) * CINPAD + lh * 16;

    auto loadA = [&](int it) -> v16h {
        const int tap = it / NCH;
        const int kc  = it - tap * NCH;
        const _Float16* ap = abase + tap * APITCH + kc * 32;
        v8h lo = *(const v8h*)ap;
        v8h hi = *(const v8h*)(ap + 8);
        return __builtin_shufflevector(lo, hi,
                 0,1,2,3,4,5,6,7,8,9,10,11,12,13,14,15);
    };
    auto loadB = [&](int it, int sp) -> v16h {
        const int tap = it / NCH;
        const int kc  = it - tap * NCH;
        const int dy  = tap / K;
        const int dx  = tap - dy * K;
        return *(const v16h*)(bbase + (dy * LDSW + dx + 16 * sp) * CINPAD + kc * 32);
    };

    v8f acc0 = {}, acc1 = {};
    v16h a0 = loadA(0);
    v16h a1 = (NIT > 1) ? loadA(1) : a0;
    v16h b0 = loadB(0, 0);
    v16h b1 = loadB(0, 1);
#pragma unroll
    for (int it = 0; it < NIT; ++it) {
        v16h an  = (it + 2 < NIT) ? loadA(it + 2) : a1;       // A: distance 2
        v16h bn0 = (it + 1 < NIT) ? loadB(it + 1, 0) : b0;    // B: distance 1
        v16h bn1 = (it + 1 < NIT) ? loadB(it + 1, 1) : b1;
        acc0 = __builtin_amdgcn_wmma_f32_16x16x32_f16(
                   false, a0, false, b0, (short)0, acc0, false, false);
        acc1 = __builtin_amdgcn_wmma_f32_16x16x32_f16(
                   false, a0, false, b1, (short)0, acc1, false, false);
        a0 = a1; a1 = an; b0 = bn0; b1 = bn1;
    }
#pragma unroll
    for (int r = 0; r < 8; ++r) {
        const int m = lh * 8 + r;                // 16x16 f32 D layout
        float* op = out + (((size_t)b * COUT + mt * 16 + m) * HW + y) * HW + lm;
        op[0]  = acc0[r];
        op[16] = acc1[r];
    }
}

template<int CIN, int CINPAD, int COUT, int K, int PAD>
static inline void convT(hipStream_t s, const float* in, const _Float16* w, float* o)
{
    constexpr int MT  = COUT / 16;
    constexpr int NGY = (MT + 7) / 8;
    k_conv<CIN, CINPAD, COUT, K, PAD><<<dim3(64, NGY), 256, 0, s>>>(in, w, o);
}

// ------------------------- fused elementwise gate kernels ------------------
__device__ __forceinline__ float sigm(float x) { return 1.0f / (1.0f + __expf(-x)); }
// read 64-ch chunk j of a [B][Cout][32][32] conv output; r in [0, 65536)
__device__ __forceinline__ float chv(const float* p, int Cout, int b, int chunk, int r) {
    return p[((size_t)b * Cout + chunk * 64) * PIX + r];
}

__global__ void k_st_gates(const float* __restrict__ xc, const float* __restrict__ hc,
                           const float* __restrict__ mc, float* __restrict__ cst,
                           float* __restrict__ mst, float* __restrict__ mem)
{
    int e = blockIdx.x * blockDim.x + threadIdx.x;
    if (e >= T64) return;
    int b = e >> 16, r = e & 65535;
    float i  = sigm (chv(xc,448,b,0,r) + chv(hc,256,b,0,r));
    float f  = sigm (chv(xc,448,b,1,r) + chv(hc,256,b,1,r) + 1.0f);
    float g  = tanhf(chv(xc,448,b,2,r) + chv(hc,256,b,2,r));
    float cn = f * cst[e] + i * g;
    float ip = sigm (chv(xc,448,b,3,r) + chv(mc,192,b,0,r));
    float fp = sigm (chv(xc,448,b,4,r) + chv(mc,192,b,1,r) + 1.0f);
    float gp = tanhf(chv(xc,448,b,5,r) + chv(mc,192,b,2,r));
    float mn = fp * mst[e] + ip * gp;
    cst[e] = cn;
    mst[e] = mn;
    mem[(size_t)b * 128 * PIX + r]        = cn;   // mem = concat(c_new, m_new)
    mem[((size_t)b * 128 + 64) * PIX + r] = mn;
}

__global__ void k_st_out(const float* __restrict__ xc, const float* __restrict__ hc,
                         const float* __restrict__ oc, const float* __restrict__ wl,
                         float* __restrict__ h)
{
    int e = blockIdx.x * blockDim.x + threadIdx.x;
    if (e >= T64) return;
    int b = e >> 16, r = e & 65535;
    float o = sigm(chv(xc,448,b,6,r) + chv(hc,256,b,3,r) + oc[e]);
    h[e] = o * tanhf(wl[e]);
}

// _gated_mem / mimn_cell: hc,xc are 4x64-ch conv outputs (order i,g,f,o)
__global__ void k_gated_mem(const float* __restrict__ hc, const float* __restrict__ xc,
                            const float* __restrict__ ctw, const float* __restrict__ ocw,
                            float* __restrict__ cst, float* __restrict__ hout)
{
    int e = blockIdx.x * blockDim.x + threadIdx.x;
    if (e >= T64) return;
    int b = e >> 16, r = e & 65535;
    float cc = cst[e];
    float i_ = sigm(chv(hc,256,b,0,r) + cc * ctw[r]         + chv(xc,256,b,0,r));
    float f_ = sigm(chv(hc,256,b,2,r) + cc * ctw[65536 + r] + chv(xc,256,b,2,r) + 1.0f);
    float cn = f_ * cc + i_ * tanhf(chv(hc,256,b,1,r) + chv(xc,256,b,1,r));
    float hn = sigm(chv(hc,256,b,3,r) + chv(xc,256,b,3,r) + cn * ocw[r]) * tanhf(cn);
    cst[e]  = cn;
    hout[e] = hn;
}

__global__ void k_mim_gates(const float* __restrict__ tcc, const float* __restrict__ scc,
                            const float* __restrict__ xcc, const float* __restrict__ cmims,
                            float* __restrict__ cst, float* __restrict__ mst,
                            float* __restrict__ mem)
{
    int e = blockIdx.x * blockDim.x + threadIdx.x;
    if (e >= T64) return;
    int b = e >> 16, r = e & 65535;
    float ix  = chv(xcc,256,b,0,r), gx = chv(xcc,256,b,1,r), fx = chv(xcc,256,b,2,r);
    float it  = chv(tcc,192,b,0,r), gt = chv(tcc,192,b,1,r);
    float is_ = chv(scc,256,b,0,r), gs = chv(scc,256,b,1,r), fs = chv(scc,256,b,2,r);
    float i  = sigm (ix + it);
    float i_ = sigm (ix + is_);
    float g  = tanhf(gx + gt);
    float g_ = tanhf(gx + gs);
    float f_ = sigm (fx + fs + 1.0f);
    float mn = f_ * mst[e] + i_ * g_;
    float cn = cmims[e] + i * g;
    cst[e] = cn;
    mst[e] = mn;
    mem[(size_t)b * 128 * PIX + r]        = cn;
    mem[((size_t)b * 128 + 64) * PIX + r] = mn;
}

__global__ void k_mim_out(const float* __restrict__ tcc, const float* __restrict__ scc,
                          const float* __restrict__ xcc, const float* __restrict__ wl,
                          float* __restrict__ h)
{
    int e = blockIdx.x * blockDim.x + threadIdx.x;
    if (e >= T64) return;
    int b = e >> 16, r = e & 65535;
    float o = sigm(chv(xcc,256,b,3,r) + chv(tcc,192,b,2,r) + chv(scc,256,b,3,r));
    h[e] = o * tanhf(wl[e]);
}

__global__ void k_diff(const float* __restrict__ a, const float* __restrict__ b,
                       float* __restrict__ d)
{
    int e = blockIdx.x * blockDim.x + threadIdx.x;
    if (e < T64) d[e] = a[e] - b[e];
}

// x_enc[B,8,128,128,1] -> x_t[B,16,32,32], channel d = py*4 + px
__global__ void k_patchify(const float* __restrict__ xenc, float* __restrict__ xt, int t)
{
    int e = blockIdx.x * blockDim.x + threadIdx.x;
    if (e >= NB * 16 * PIX) return;
    int j = e & 31, i = (e >> 5) & 31, d = (e >> 10) & 15, b = e >> 14;
    int py = d >> 2, px = d & 3;
    xt[e] = xenc[(((size_t)b * 8 + t) * 128 + (py * 32 + i)) * 128 + (px * 32 + j)];
}

// x_gen[B,16,32,32] -> out[B,8,128,128,1] frame p
__global__ void k_unpatchify(const float* __restrict__ xg, float* __restrict__ out, int p)
{
    int e = blockIdx.x * blockDim.x + threadIdx.x;
    if (e >= NB * 16 * PIX) return;
    int j = e & 31, i = (e >> 5) & 31, d = (e >> 10) & 15, b = e >> 14;
    int py = d >> 2, px = d & 3;
    out[(((size_t)b * 8 + p) * 128 + (py * 32 + i)) * 128 + (px * 32 + j)] = xg[e];
}

// ---------------------------------------------------------------------------
static inline char* bump(char*& p, size_t bytes)
{
    char* r = p;
    p += (bytes + 255) & ~(size_t)255;
    return r;
}

extern "C" void kernel_launch(void* const* d_in, const int* in_sizes, int n_in,
                              void* d_out, int out_size, void* d_ws, size_t ws_size,
                              hipStream_t stream)
{
    (void)in_sizes; (void)n_in; (void)out_size; (void)ws_size;
    const float* xenc = (const float*)d_in[0];
    float* out = (float*)d_out;

    // Input dict order: x_enc, x_mark_enc, x_dec, x_mark_dec, then params:
    // cell0{wx,wh,wm,wo,wlast}=4..8, w_last=9,
    // per i in 1..3: mim_i{wt,ws,wx,mims_wh,mims_wx,ct_w,oc_w,wlast}=base..base+7,
    //               mimn_i{wh,wx,ct_w,oc_w}=base+8..base+11, base = 10+(i-1)*12.

    char* p = (char*)d_ws;
    auto fbuf = [&](size_t elems) -> float*    { return (float*)bump(p, elems * 4); };
    auto hbuf = [&](size_t elems) -> _Float16* { return (_Float16*)bump(p, elems * 2); };
    auto TN   = [](int ch) -> size_t { return (size_t)NB * ch * PIX; };

    // ---- persistent state ----
    float *h[4], *c[4], *dh[3], *dc[3], *clc[3];
    for (int i = 0; i < 4; ++i) h[i]   = fbuf(TN(64));
    for (int i = 0; i < 4; ++i) c[i]   = fbuf(TN(64));
    float* st   = fbuf(TN(64));
    for (int i = 0; i < 3; ++i) dh[i]  = fbuf(TN(64));
    for (int i = 0; i < 3; ++i) dc[i]  = fbuf(TN(64));
    for (int i = 0; i < 3; ++i) clc[i] = fbuf(TN(64));
    float* preh  = fbuf(TN(64));
    float* diffb = fbuf(TN(64));
    float* zbuf  = fbuf(TN(64));
    float* x_t   = fbuf(TN(16));
    float* x_gen = fbuf(TN(16));

    // ---- temporaries ----
    float* t_xc  = fbuf(TN(448));
    float* t_hc  = fbuf(TN(256));
    float* t_mc  = fbuf(TN(192));
    float* t_mem = fbuf(TN(128));
    float* t_oc  = fbuf(TN(64));
    float* t_wl  = fbuf(TN(64));
    float* t_cm  = fbuf(TN(64));
    float* t_tcc = fbuf(TN(192));
    float* t_scc = fbuf(TN(256));
    float* t_xcc = fbuf(TN(256));
    float* t_h2  = fbuf(TN(256));
    float* t_x2  = fbuf(TN(256));

    // ---- f16 repacked weights ----
    _Float16* w_c0_wx    = hbuf((size_t)25 * 448 * 32);
    _Float16* w_c0_wh    = hbuf((size_t)25 * 256 * 64);
    _Float16* w_c0_wm    = hbuf((size_t)25 * 192 * 64);
    _Float16* w_c0_wo    = hbuf((size_t)25 * 64 * 128);
    _Float16* w_c0_wlast = hbuf((size_t)64 * 128);
    _Float16* w_last     = hbuf((size_t)16 * 64);
    _Float16 *w_mim_wt[3], *w_mim_ws[3], *w_mim_wx[3], *w_mim_wh2[3], *w_mim_wx2[3],
             *w_mim_wlast[3], *w_mimn_wh[3], *w_mimn_wx[3];
    for (int i = 0; i < 3; ++i) {
        w_mim_wt[i]    = hbuf((size_t)25 * 192 * 64);
        w_mim_ws[i]    = hbuf((size_t)25 * 256 * 64);
        w_mim_wx[i]    = hbuf((size_t)25 * 256 * 64);
        w_mim_wh2[i]   = hbuf((size_t)25 * 256 * 64);
        w_mim_wx2[i]   = hbuf((size_t)25 * 256 * 64);
        w_mim_wlast[i] = hbuf((size_t)64 * 128);
        w_mimn_wh[i]   = hbuf((size_t)25 * 256 * 64);
        w_mimn_wx[i]   = hbuf((size_t)25 * 256 * 64);
    }

    auto cvt = [&](const void* src, _Float16* dst, int Cout, int Cin, int CinPad, int taps) {
        int total = taps * Cout * CinPad;
        k_convert_w<<<(total + 255) / 256, 256, 0, stream>>>(
            (const float*)src, dst, Cout, Cin, CinPad, taps);
    };
    cvt(d_in[4], w_c0_wx,    448,  16,  32, 25);
    cvt(d_in[5], w_c0_wh,    256,  64,  64, 25);
    cvt(d_in[6], w_c0_wm,    192,  64,  64, 25);
    cvt(d_in[7], w_c0_wo,     64, 128, 128, 25);
    cvt(d_in[8], w_c0_wlast,  64, 128, 128,  1);
    cvt(d_in[9], w_last,      16,  64,  64,  1);
    for (int i = 0; i < 3; ++i) {
        int base = 10 + i * 12;
        cvt(d_in[base + 0], w_mim_wt[i],    192, 64, 64, 25);
        cvt(d_in[base + 1], w_mim_ws[i],    256, 64, 64, 25);
        cvt(d_in[base + 2], w_mim_wx[i],    256, 64, 64, 25);
        cvt(d_in[base + 3], w_mim_wh2[i],   256, 64, 64, 25);
        cvt(d_in[base + 4], w_mim_wx2[i],   256, 64, 64, 25);
        cvt(d_in[base + 7], w_mim_wlast[i],  64, 128, 128, 1);
        cvt(d_in[base + 8], w_mimn_wh[i],   256, 64, 64, 25);
        cvt(d_in[base + 9], w_mimn_wx[i],   256, 64, 64, 25);
    }

    // ---- zero initial state ----
    const size_t b64 = TN(64) * 4;
    for (int i = 0; i < 4; ++i) hipMemsetAsync(h[i], 0, b64, stream);
    for (int i = 0; i < 4; ++i) hipMemsetAsync(c[i], 0, b64, stream);
    hipMemsetAsync(st, 0, b64, stream);
    for (int i = 0; i < 3; ++i) hipMemsetAsync(dh[i],  0, b64, stream);
    for (int i = 0; i < 3; ++i) hipMemsetAsync(dc[i],  0, b64, stream);
    for (int i = 0; i < 3; ++i) hipMemsetAsync(clc[i], 0, b64, stream);
    hipMemsetAsync(zbuf, 0, b64, stream);

    auto stlstm = [&](const float* xt) {
        convT<16,  32, 448, 5, 2>(stream, xt,    w_c0_wx, t_xc);
        convT<64,  64, 256, 5, 2>(stream, h[0],  w_c0_wh, t_hc);
        convT<64,  64, 192, 5, 2>(stream, st,    w_c0_wm, t_mc);
        k_st_gates<<<512, 256, 0, stream>>>(t_xc, t_hc, t_mc, c[0], st, t_mem);
        convT<128, 128, 64, 5, 2>(stream, t_mem, w_c0_wo,    t_oc);
        convT<128, 128, 64, 1, 0>(stream, t_mem, w_c0_wlast, t_wl);
        k_st_out<<<512, 256, 0, stream>>>(t_xc, t_hc, t_oc, t_wl, h[0]);
    };

    auto mimn = [&](int i, const float* x) {    // i in 1..3
        int base = 10 + (i - 1) * 12;
        convT<64, 64, 256, 5, 2>(stream, dh[i - 1], w_mimn_wh[i - 1], t_h2);
        convT<64, 64, 256, 5, 2>(stream, x,         w_mimn_wx[i - 1], t_x2);
        k_gated_mem<<<512, 256, 0, stream>>>(t_h2, t_x2,
            (const float*)d_in[base + 10], (const float*)d_in[base + 11],
            dc[i - 1], dh[i - 1]);
    };

    auto mim = [&](int i, const float* diffh) { // i in 1..3
        int base = 10 + (i - 1) * 12;
        convT<64, 64, 192, 5, 2>(stream, h[i],     w_mim_wt[i - 1],  t_tcc);
        convT<64, 64, 256, 5, 2>(stream, st,       w_mim_ws[i - 1],  t_scc);
        convT<64, 64, 256, 5, 2>(stream, h[i - 1], w_mim_wx[i - 1],  t_xcc);
        convT<64, 64, 256, 5, 2>(stream, c[i],     w_mim_wh2[i - 1], t_h2);
        convT<64, 64, 256, 5, 2>(stream, diffh,    w_mim_wx2[i - 1], t_x2);
        k_gated_mem<<<512, 256, 0, stream>>>(t_h2, t_x2,
            (const float*)d_in[base + 5], (const float*)d_in[base + 6],
            clc[i - 1], t_cm);
        k_mim_gates<<<512, 256, 0, stream>>>(t_tcc, t_scc, t_xcc, t_cm, c[i], st, t_mem);
        convT<128, 128, 64, 1, 0>(stream, t_mem, w_mim_wlast[i - 1], t_wl);
        k_mim_out<<<512, 256, 0, stream>>>(t_tcc, t_scc, t_xcc, t_wl, h[i]);
    };

    // ---- t = 0 ----
    k_patchify<<<128, 256, 0, stream>>>(xenc, x_t, 0);
    stlstm(x_t);
    for (int i = 1; i < 4; ++i) mim(i, zbuf);
    convT<64, 64, 16, 1, 0>(stream, h[3], w_last, x_gen);

    // ---- t = 1..14 (scan); predictions emitted for t >= 7 ----
    for (int t = 1; t < 15; ++t) {
        if (t < 8)
            k_patchify<<<128, 256, 0, stream>>>(xenc, x_t, t);
        else
            hipMemcpyAsync(x_t, x_gen, TN(16) * 4, hipMemcpyDeviceToDevice, stream);
        hipMemcpyAsync(preh, h[0], b64, hipMemcpyDeviceToDevice, stream);
        stlstm(x_t);
        k_diff<<<512, 256, 0, stream>>>(h[0], preh, diffb);
        mimn(1, diffb);
        mim(1, dh[0]);
        mimn(2, dh[0]);
        mim(2, dh[1]);
        mimn(3, dh[1]);
        mim(3, dh[2]);
        convT<64, 64, 16, 1, 0>(stream, h[3], w_last, x_gen);
        if (t >= 7)
            k_unpatchify<<<128, 256, 0, stream>>>(x_gen, out, t - 7);
    }
}
